// FixedTemporalSpectralGNN_85048942396198
// MI455X (gfx1250) — matrix operand order repaired and checked by reading, not compiled, not run on hardware
//
#include <hip/hip_runtime.h>
#include <hip/hip_bf16.h>

#define NN 32768
#define HH 256
#define EE 262144
#define ETOT (EE + NN)
#define BB 32

typedef __attribute__((ext_vector_type(16))) __bf16 v16bf;
typedef __attribute__((ext_vector_type(8)))  float  v8f;

// ---------------------------------------------------------------- helpers ---
__device__ __forceinline__ unsigned short f2bf(float x) {
    unsigned u = __float_as_uint(x);
    u += 0x7FFFu + ((u >> 16) & 1u);          // round-to-nearest-even
    return (unsigned short)(u >> 16);
}
__device__ __forceinline__ float eluf(float x) { return x > 0.f ? x : __expf(x) - 1.f; }
__device__ __forceinline__ void atomicMaxF(float* addr, float val) {
    int* ai = (int*)addr;
    int old = *ai;
    while (__int_as_float(old) < val) {
        int assumed = old;
        old = atomicCAS(ai, assumed, __float_as_int(val));
        if (old == assumed) break;
    }
}

// ------------------------------------------------------- generic WMMA GEMM --
// C[m,n] = sum_k A[m,k]*Wt[n,k] + bias[n].  A f32 (lda), Wt bf16 pairs as uint
// ([Nout][K], pair = K even|K odd), C f32 (ldc). M%128==0, Nout%128==0, K%32==0.
// B tile staged via GLOBAL_LOAD_ASYNC_TO_LDS_B128 (ASYNCcnt), A tile converted
// f32->bf16 on VALU with a global_prefetch of the next K-step.
__global__ __launch_bounds__(256)
void k_gemm_bf16(const float* __restrict__ A, int lda,
                 const unsigned* __restrict__ Wt,
                 const float* __restrict__ bias,
                 float* __restrict__ C, int ldc, int K) {
    __shared__ __align__(16) unsigned Abuf[128 * 16];
    __shared__ __align__(16) unsigned Bbuf[128 * 16];
    const int tid   = threadIdx.x;
    const int lane  = tid & 31;
    const int wave  = tid >> 5;
    const int rowg  = wave & 3;        // 4 row groups of 32
    const int colg  = wave >> 2;       // 2 col groups of 64
    const int half  = lane >> 4;
    const int l16   = lane & 15;
    const int rowB  = blockIdx.y * 128;
    const int colB  = blockIdx.x * 128;
    const int Kp    = K >> 1;

    v8f acc[2][4];
#pragma unroll
    for (int a = 0; a < 2; a++)
#pragma unroll
        for (int b = 0; b < 4; b++)
#pragma unroll
            for (int i = 0; i < 8; i++) acc[a][b][i] = 0.f;

    for (int k0 = 0; k0 < K; k0 += 32) {
        // --- async DMA: B tile (bf16 weights) straight into LDS, 16B/lane ---
#pragma unroll
        for (int it = 0; it < 2; it++) {
            int u = it * 1024 + tid * 4;              // uint index in Bbuf
            int n = u >> 4, i = u & 15;               // row, pair within row
            unsigned long long ga = (unsigned long long)(const void*)
                (Wt + (size_t)(colB + n) * Kp + (k0 >> 1) + i);
            unsigned ldsa = (unsigned)(unsigned long long)(&Bbuf[u]);
            asm volatile("global_load_async_to_lds_b128 %0, %1, off"
                         :: "v"(ldsa), "v"(ga) : "memory");
        }
        // --- A tile: f32 -> bf16 on VALU ---
#pragma unroll
        for (int it = 0; it < 8; it++) {
            int idx = tid + it * 256;          // 0..2047
            int r = idx >> 4, i = idx & 15;
            const float* ap = A + (size_t)(rowB + r) * lda + k0 + 2 * i;
            unsigned lo = f2bf(ap[0]);
            unsigned hi = f2bf(ap[1]);
            Abuf[idx] = lo | (hi << 16);
            if (k0 + 32 < K) __builtin_prefetch(ap + 32, 0, 3);
        }
        asm volatile("s_wait_asynccnt 0x0" ::: "memory");
        __syncthreads();

        union Frag { unsigned u[8]; v16bf v; };
        Frag bf[4];
#pragma unroll
        for (int ct = 0; ct < 4; ct++) {
            int n = colg * 64 + ct * 16 + l16;
#pragma unroll
            for (int j = 0; j < 8; j++) bf[ct].u[j] = Bbuf[n * 16 + half * 8 + j];
        }
#pragma unroll
        for (int rt = 0; rt < 2; rt++) {
            int m = rowg * 32 + rt * 16 + l16;
            Frag af;
#pragma unroll
            for (int j = 0; j < 8; j++) {
                int i = half * 8 + j;
                af.u[j] = Abuf[m * 16 + i];
            }
#pragma unroll
            for (int ct = 0; ct < 4; ct++) {
                acc[rt][ct] = __builtin_amdgcn_wmma_f32_16x16x32_bf16(
                    false, af.v, false, bf[ct].v, (short)0, acc[rt][ct], false, false);
            }
        }
        __syncthreads();
    }

#pragma unroll
    for (int ct = 0; ct < 4; ct++) {
        int col = colB + colg * 64 + ct * 16 + l16;
        float bv = bias ? bias[col] : 0.f;
#pragma unroll
        for (int rt = 0; rt < 2; rt++) {
#pragma unroll
            for (int v = 0; v < 8; v++) {
                int row = rowB + rowg * 32 + rt * 16 + v + half * 8;
                C[(size_t)row * ldc + col] = acc[rt][ct][v] + bv;
            }
        }
    }
}

// ---------------------------------------------------- weight f32 -> bf16^T --
__global__ void k_convw(const float* __restrict__ W, unsigned short* __restrict__ Wt,
                        int K, int Nout) {
    int idx = blockIdx.x * 256 + threadIdx.x;
    if (idx >= K * Nout) return;
    int n = idx / K, k = idx - n * K;
    Wt[idx] = f2bf(W[(size_t)k * Nout + n]);
}

// ---------------------------------------------------------------- fill ------
__global__ void k_fill(float* p, float v, size_t n) {
    size_t i = (size_t)blockIdx.x * 256 + threadIdx.x;
    if (i < n) p[i] = v;
}

// ----------------------------------------------------- row LayerNorm (D<=512)
__global__ __launch_bounds__(256)
void k_ln(const float* __restrict__ X, const float* __restrict__ R,
          const float* __restrict__ g, const float* __restrict__ bt,
          float* __restrict__ Y, int rows, int D, int preElu, int postElu) {
    int row  = blockIdx.x * 8 + (threadIdx.x >> 5);
    int lane = threadIdx.x & 31;
    if (row >= rows) return;
    int per = D >> 5;
    float v[16];
    float s = 0.f;
    for (int i = 0; i < per; i++) {
        size_t o = (size_t)row * D + i * 32 + lane;
        float x = X[o];
        if (R) x += R[o];
        if (preElu) x = eluf(x);
        v[i] = x; s += x;
    }
    for (int o = 16; o > 0; o >>= 1) s += __shfl_xor(s, o, 32);
    float mean = s / D;
    float var = 0.f;
    for (int i = 0; i < per; i++) { float d = v[i] - mean; var += d * d; }
    for (int o = 16; o > 0; o >>= 1) var += __shfl_xor(var, o, 32);
    float rstd = rsqrtf(var / D + 1e-5f);
    for (int i = 0; i < per; i++) {
        int c = i * 32 + lane;
        float y = (v[i] - mean) * rstd * g[c] + bt[c];
        if (postElu) y = eluf(y);
        Y[(size_t)row * D + c] = y;
    }
}

// --------------------------------------------------------- spectral stage ---
__device__ __forceinline__ void ln32(float* v, const float* g, const float* b) {
    float m = 0.f;
    for (int i = 0; i < 32; i++) m += v[i];
    m *= (1.f / 32.f);
    float var = 0.f;
    for (int i = 0; i < 32; i++) { float d = v[i] - m; var += d * d; }
    var *= (1.f / 32.f);
    float r = rsqrtf(var + 1e-5f);
    for (int i = 0; i < 32; i++) v[i] = (v[i] - m) * r * g[i] + b[i];
}

__global__ __launch_bounds__(64)
void k_spec_tokens(const float* __restrict__ eigvals, const unsigned char* __restrict__ mask,
                   const float* e1w, const float* e1b, const float* l1g, const float* l1b,
                   const float* e2w, const float* e2b, const float* l2g, const float* l2b,
                   const float* wq, const float* bq, const float* wk, const float* bk,
                   const float* wv, const float* bv, const float* wo, const float* bo,
                   const float* f1w, const float* f1b, const float* f2w, const float* f2b,
                   float* __restrict__ filt) {
    __shared__ float Km[16][32], Vm[16][32];
    int b = blockIdx.x, t = threadIdx.x;
    float F[32], Q[32];
    if (t < 16) {
        float ev = eigvals[b * 16 + t];
        float h1[32];
        for (int j = 0; j < 32; j++) h1[j] = ev * e1w[j] + e1b[j];
        ln32(h1, l1g, l1b);
        for (int j = 0; j < 32; j++) h1[j] = fmaxf(h1[j], 0.f);
        for (int j = 0; j < 32; j++) {
            float s = e2b[j];
            for (int i = 0; i < 32; i++) s += h1[i] * e2w[i * 32 + j];
            F[j] = s;
        }
        ln32(F, l2g, l2b);
        for (int j = 0; j < 32; j++) {
            float sq = bq[j], sk = bk[j], sv = bv[j];
            for (int i = 0; i < 32; i++) {
                float f = F[i];
                sq += f * wq[i * 32 + j];
                sk += f * wk[i * 32 + j];
                sv += f * wv[i * 32 + j];
            }
            Q[j] = sq; Km[t][j] = sk; Vm[t][j] = sv;
        }
    }
    __syncthreads();
    if (t < 16) {
        float ctx[32];
        for (int h = 0; h < 4; h++) {
            float sc[16], mx = -1e30f;
            for (int kk = 0; kk < 16; kk++) {
                float s = 0.f;
                for (int d = 0; d < 8; d++) s += Q[h * 8 + d] * Km[kk][h * 8 + d];
                s *= 0.35355339059f;                 // 1/sqrt(8)
                if (!mask[b * 16 + kk]) s = -1e9f;
                sc[kk] = s; mx = fmaxf(mx, s);
            }
            float den = 0.f;
            for (int kk = 0; kk < 16; kk++) { float e = __expf(sc[kk] - mx); sc[kk] = e; den += e; }
            float rden = 1.f / den;
            for (int d = 0; d < 8; d++) {
                float o = 0.f;
                for (int kk = 0; kk < 16; kk++) o += sc[kk] * Vm[kk][h * 8 + d];
                ctx[h * 8 + d] = o * rden;
            }
        }
        float c2[32];
        for (int j = 0; j < 32; j++) {
            float s = bo[j];
            for (int i = 0; i < 32; i++) s += ctx[i] * wo[i * 32 + j];
            c2[j] = s;
        }
        float acc = f2b[0];
        for (int j = 0; j < 64; j++) {
            float u = f1b[j];
            for (int i = 0; i < 32; i++) u += c2[i] * f1w[i * 64 + j];
            u = fmaxf(u, 0.f);
            acc += u * f2w[j];
        }
        float fv = tanhf(acc);
        if (!mask[b * 16 + t]) fv = 0.f;
        filt[b * 16 + t] = fv;
    }
}

__global__ __launch_bounds__(256)
void k_spec_xf(const float* __restrict__ x, const float* __restrict__ ev,
               const float* __restrict__ filt, float* __restrict__ fxf) {
    int b = blockIdx.x;
    int d = threadIdx.x & 127;
    int k0 = threadIdx.x >> 7;
    float acc[8];
#pragma unroll
    for (int i = 0; i < 8; i++) acc[i] = 0.f;
    for (int n = 0; n < 1024; n++) {
        size_t node = (size_t)b * 1024 + n;
        float xv = x[node * 128 + d];
#pragma unroll
        for (int i = 0; i < 8; i++) acc[i] += ev[node * 16 + k0 + 2 * i] * xv;
    }
#pragma unroll
    for (int i = 0; i < 8; i++) {
        int k = k0 + 2 * i;
        fxf[((size_t)b * 16 + k) * 128 + d] = acc[i] * filt[b * 16 + k];
    }
}

__global__ __launch_bounds__(256)
void k_spec_xs(const float* __restrict__ ev, const float* __restrict__ fxf,
               float* __restrict__ xs) {
    size_t idx = (size_t)blockIdx.x * 256 + threadIdx.x;
    int node = (int)(idx >> 7), d = (int)(idx & 127);
    int b = node >> 10;
    float s = 0.f;
#pragma unroll
    for (int k = 0; k < 16; k++) s += ev[(size_t)node * 16 + k] * fxf[((size_t)b * 16 + k) * 128 + d];
    xs[idx] = s;
}

// --------------------------------------------------------- temporal stage ---
__global__ __launch_bounds__(256)
void k_temporal(const float* __restrict__ mask, const int* __restrict__ steps,
                const float* __restrict__ tab, const float* __restrict__ hs,
                float* __restrict__ z) {
    int n = blockIdx.x, d = threadIdx.x;
    int st = steps[n];
    int i1 = min(max(st, 0), 99);
    int i2 = min(max(st / 10, 0), 99);
    int i3 = min(max(st / 100, 0), 99);
    float e = tab[(0 * 100 + i1) * 256 + d] + tab[(1 * 100 + i2) * 256 + d] +
              tab[(2 * 100 + i3) * 256 + d];
    z[(size_t)n * 512 + d]       = e * mask[n];
    z[(size_t)n * 512 + 256 + d] = hs[(size_t)n * 256 + d];
}

// ----------------------------------------------------------- struct stage ---
__global__ void k_edge_attr(const int* __restrict__ ei, const int* __restrict__ ea,
                            const float* __restrict__ emb, float* __restrict__ lat,
                            float* __restrict__ cnt) {
    size_t idx = (size_t)blockIdx.x * 256 + threadIdx.x;
    if (idx >= (size_t)EE * 64) return;
    int e = (int)(idx >> 6), j = (int)(idx & 63);
    int dst = ei[EE + e];
    int a = ea[e];
    atomicAdd(&lat[(size_t)dst * 64 + j], emb[a * 64 + j]);
    if (j == 0) atomicAdd(&cnt[dst], 1.0f);
}

__global__ void k_loop_div(float* __restrict__ lat, const float* __restrict__ cnt) {
    size_t idx = (size_t)blockIdx.x * 256 + threadIdx.x;
    if (idx >= (size_t)NN * 64) return;
    lat[idx] /= fmaxf(cnt[idx >> 6], 1.0f);
}

__global__ __launch_bounds__(256)
void k_ee3(const float* __restrict__ emb, const float* __restrict__ we,
           float* __restrict__ ee3) {
    int d = threadIdx.x;
    for (int r = 0; r < 3; r++) {
        float s = 0.f;
        for (int j = 0; j < 64; j++) s += emb[r * 64 + j] * we[(size_t)j * 256 + d];
        ee3[r * 256 + d] = s;
    }
}

__global__ __launch_bounds__(256)
void k_gat_score(const int* __restrict__ ei, const int* __restrict__ ea,
                 const float* __restrict__ xl, const float* __restrict__ xr,
                 const float* __restrict__ ee3, const float* __restrict__ eel,
                 const float* __restrict__ att, float* __restrict__ esc,
                 float* __restrict__ nmax) {
    int w = (blockIdx.x * 256 + threadIdx.x) >> 5;
    if (w >= ETOT) return;
    int lane = threadIdx.x & 31;
    int src, dst; const float* ee;
    if (w < EE) { src = ei[w]; dst = ei[EE + w]; ee = ee3 + (size_t)ea[w] * 256; }
    else        { src = dst = w - EE;            ee = eel + (size_t)(w - EE) * 256; }
    float p = 0.f;
    int d0 = lane * 8;
#pragma unroll
    for (int i = 0; i < 8; i++) {
        int d = d0 + i;
        float m = xl[(size_t)src * 256 + d] + xr[(size_t)dst * 256 + d] + ee[d];
        float lr = m > 0.f ? m : 0.2f * m;
        p += lr * att[d];
    }
    p += __shfl_xor(p, 1, 32);
    p += __shfl_xor(p, 2, 32);
    p += __shfl_xor(p, 4, 32);
    if ((lane & 7) == 0) {
        int h = lane >> 3;
        esc[(size_t)w * 4 + h] = p;
        atomicMaxF(&nmax[(size_t)dst * 4 + h], p);
    }
}

__global__ void k_gat_expsum(const int* __restrict__ ei, float* __restrict__ esc,
                             const float* __restrict__ nmax, float* __restrict__ nsum) {
    size_t idx = (size_t)blockIdx.x * 256 + threadIdx.x;
    if (idx >= (size_t)ETOT * 4) return;
    int e = (int)(idx >> 2), h = (int)(idx & 3);
    int dst = (e < EE) ? ei[EE + e] : (e - EE);
    float ex = __expf(esc[idx] - nmax[(size_t)dst * 4 + h]);
    esc[idx] = ex;
    atomicAdd(&nsum[(size_t)dst * 4 + h], ex);
}

__global__ __launch_bounds__(256)
void k_gat_aggr(const int* __restrict__ ei, const float* __restrict__ xl,
                const float* __restrict__ esc, const float* __restrict__ nsum,
                float* __restrict__ hsum) {
    int w = (blockIdx.x * 256 + threadIdx.x) >> 5;
    if (w >= ETOT) return;
    int lane = threadIdx.x & 31;
    int src, dst;
    if (w < EE) { src = ei[w]; dst = ei[EE + w]; }
    else        { src = dst = w - EE; }
    int h = lane >> 3;
    float alpha = esc[(size_t)w * 4 + h] / (nsum[(size_t)dst * 4 + h] + 1e-16f);
    int d0 = lane * 8;
#pragma unroll
    for (int i = 0; i < 8; i++) {
        int d = d0 + i;
        atomicAdd(&hsum[(size_t)dst * 256 + d], alpha * xl[(size_t)src * 256 + d]);
    }
}

__global__ void k_comb_prep(const float* __restrict__ hsum, const float* __restrict__ b1,
                            const float* __restrict__ b2, const float* __restrict__ x,
                            const float* __restrict__ tw, const float* __restrict__ tb,
                            float* __restrict__ cat) {
    size_t idx = (size_t)blockIdx.x * 256 + threadIdx.x;
    if (idx >= (size_t)NN * 320) return;
    int n = (int)(idx / 320), c = (int)(idx - (size_t)n * 320);
    float v;
    if (c < 256) v = hsum[(size_t)n * 256 + c] + b1[c] + b2[c];
    else {
        int j = c - 256;
        v = x[(size_t)n * 128 + 0] * tw[j] + x[(size_t)n * 128 + 1] * tw[64 + j] + tb[j];
    }
    cat[idx] = v;
}

// ----------------------------------------------------------- fusion stage ---
__global__ void k_concat(const float* __restrict__ hs, const float* __restrict__ h,
                         const float* __restrict__ ht, float* __restrict__ hc) {
    size_t idx = (size_t)blockIdx.x * 256 + threadIdx.x;
    if (idx >= (size_t)NN * 768) return;
    int n = (int)(idx / 768), c = (int)(idx - (size_t)n * 768);
    const float* s = (c < 256) ? hs : ((c < 512) ? h : ht);
    hc[idx] = s[(size_t)n * 256 + (c & 255)];
}

__global__ __launch_bounds__(256)
void k_scores(const float* __restrict__ z, const float* __restrict__ w,
              const float* __restrict__ b, float* __restrict__ out) {
    int row  = blockIdx.x * 8 + (threadIdx.x >> 5);
    int lane = threadIdx.x & 31;
    float s = 0.f;
#pragma unroll
    for (int i = 0; i < 8; i++) { int d = i * 32 + lane; s += z[(size_t)row * 256 + d] * w[d]; }
    for (int o = 16; o > 0; o >>= 1) s += __shfl_xor(s, o, 32);
    if (lane == 0) out[row] = s + b[0];
}

__global__ __launch_bounds__(256)
void k_value(const float* __restrict__ hc, const float* __restrict__ v1w,
             const float* __restrict__ v1b, const float* __restrict__ v2w,
             const float* __restrict__ v2b, float* __restrict__ vout) {
    __shared__ float gemb[768];
    __shared__ float u[256];
    int b = blockIdx.x, t = threadIdx.x;
    for (int c = t; c < 768; c += 256) {
        float s = 0.f;
        const float* p = hc + (size_t)b * 1024 * 768 + c;
        for (int n = 0; n < 1024; n++) s += p[(size_t)n * 768];
        gemb[c] = s * (1.f / 1024.f);
    }
    __syncthreads();
    {
        float s = v1b[t];
        for (int c = 0; c < 768; c++) s += gemb[c] * v1w[(size_t)c * 256 + t];
        u[t] = eluf(s);
    }
    __syncthreads();
    if (t == 0) {
        float s = v2b[0];
        for (int i = 0; i < 256; i++) s += u[i] * v2w[i];
        vout[b] = 1.f / (1.f + __expf(-s));
    }
}

// ================================================================= host =====
static void conv(const void* W, unsigned short* dst, int K, int Nout, hipStream_t s) {
    int total = K * Nout;
    k_convw<<<(total + 255) / 256, 256, 0, s>>>((const float*)W, dst, K, Nout);
}
static void gemm(const float* A, int lda, const unsigned short* Wt, const float* bias,
                 float* C, int ldc, int M, int K, int Nout, hipStream_t s) {
    dim3 g(Nout / 128, M / 128);
    k_gemm_bf16<<<g, 256, 0, s>>>(A, lda, (const unsigned*)Wt, bias, C, ldc, K);
}
static void fill(float* p, float v, size_t n, hipStream_t s) {
    k_fill<<<(unsigned)((n + 255) / 256), 256, 0, s>>>(p, v, n);
}
static void ln(const float* X, const float* R, const float* g, const float* b, float* Y,
               int rows, int D, int pre, int post, hipStream_t s) {
    k_ln<<<(rows + 7) / 8, 256, 0, s>>>(X, R, g, b, Y, rows, D, pre, post);
}

// workspace layout (float units)
static constexpr size_t O_HS   = 0;
static constexpr size_t O_H    = O_HS   + (size_t)NN * 256;
static constexpr size_t O_HT   = O_H    + (size_t)NN * 256;
static constexpr size_t O_XL   = O_HT   + (size_t)NN * 256;
static constexpr size_t O_XR   = O_XL   + (size_t)NN * 256;
static constexpr size_t O_HSUM = O_XR   + (size_t)NN * 256;
static constexpr size_t O_HNEW = O_HSUM + (size_t)NN * 256;
static constexpr size_t O_CAT  = O_HNEW + (size_t)NN * 256;
static constexpr size_t O_Z512 = O_CAT  + (size_t)NN * 320;
static constexpr size_t O_Z256 = O_Z512 + (size_t)NN * 512;
static constexpr size_t O_LAT  = O_Z256 + (size_t)NN * 256;
static constexpr size_t O_EEL  = O_LAT  + (size_t)NN * 64;
static constexpr size_t O_CNT  = O_EEL  + (size_t)NN * 256;
static constexpr size_t O_ESC  = O_CNT  + NN;
static constexpr size_t O_NMAX = O_ESC  + (size_t)ETOT * 4;
static constexpr size_t O_NSUM = O_NMAX + (size_t)NN * 4;
static constexpr size_t O_FILT = O_NSUM + (size_t)NN * 4;
static constexpr size_t O_FXF  = O_FILT + 512;
static constexpr size_t O_XS   = O_FXF  + 32 * 16 * 128;
static constexpr size_t O_EE3  = O_XS   + (size_t)NN * 128;
static constexpr size_t O_BF   = O_EE3  + 768;          // bf16 region (ushort units below)

static constexpr size_t W_SPECOUT = 0;
static constexpr size_t W_TPROJ   = W_SPECOUT + 128 * 256;
static constexpr size_t W_INPROJ  = W_TPROJ   + 512 * 256;
static constexpr size_t GAT_SZ    = 64 * 256 + 256 * 256 + 256 * 256;
static constexpr size_t LAYER_SZ  = 2 * GAT_SZ + 320 * 256;
static constexpr size_t W_LAYER   = W_INPROJ + 128 * 256;
static constexpr size_t W_F1      = W_LAYER + 3 * LAYER_SZ;
static constexpr size_t W_F2      = W_F1 + 768 * 512;

extern "C" void kernel_launch(void* const* d_in, const int* in_sizes, int n_in,
                              void* d_out, int out_size, void* d_ws, size_t ws_size,
                              hipStream_t stream) {
    (void)in_sizes; (void)n_in; (void)out_size; (void)ws_size;
#define F(i) ((const float*)d_in[(i)])
#define I(i) ((const int*)d_in[(i)])
    float* ws = (float*)d_ws;
    unsigned short* wbf = (unsigned short*)(ws + O_BF);
    float* out = (float*)d_out;
    float* hc  = out + NN;                       // N x 768 block of d_out
    float* vout = out + NN + (size_t)NN * 768;

    // ---- weight conversion to bf16^T ----
    conv(F(40),  wbf + W_SPECOUT, 128, 256, stream);   // spec.out.w
    conv(F(119), wbf + W_TPROJ,   512, 256, stream);   // temp.proj.w
    conv(F(49),  wbf + W_INPROJ,  128, 256, stream);   // struct.in_proj.w
    for (int l = 0; l < 3; l++) {
        int base = 50 + l * 20;
        for (int g = 0; g < 2; g++) {
            int gb = base + 2 + g * 7;                 // att,bias,bl,br,we,wl,wr
            size_t wb = W_LAYER + l * LAYER_SZ + g * GAT_SZ;
            conv(F(gb + 4), wbf + wb,                       64,  256, stream); // we
            conv(F(gb + 5), wbf + wb + 64 * 256,            256, 256, stream); // wl
            conv(F(gb + 6), wbf + wb + 64 * 256 + 256 * 256,256, 256, stream); // wr
        }
        conv(F(base + 1), wbf + W_LAYER + l * LAYER_SZ + 2 * GAT_SZ, 320, 256, stream); // comb
    }
    conv(F(10), wbf + W_F1, 768, 512, stream);         // fusion.f1.w
    conv(F(12), wbf + W_F2, 512, 256, stream);         // fusion.f2.w

    // ---- spectral ----
    k_spec_tokens<<<BB, 64, 0, stream>>>(
        F(5), (const unsigned char*)d_in[6],
        F(28), F(27), F(36), F(35), F(30), F(29), F(38), F(37),
        F(25), F(21), F(23), F(19), F(26), F(22), F(24), F(20),
        F(32), F(31), F(34), F(33), ws + O_FILT);
    k_spec_xf<<<BB, 256, 0, stream>>>(F(0), F(4), ws + O_FILT, ws + O_FXF);
    k_spec_xs<<<NN * 128 / 256, 256, 0, stream>>>(F(4), ws + O_FXF, ws + O_XS);
    gemm(ws + O_XS, 128, wbf + W_SPECOUT, F(39), ws + O_HS, 256, NN, 128, 256, stream);
    ln(ws + O_HS, nullptr, F(42), F(41), ws + O_HS, NN, 256, 0, 0, stream);

    // ---- temporal (needs hs) ----
    k_temporal<<<NN, 256, 0, stream>>>(F(2), I(3), F(120), ws + O_HS, ws + O_Z512);
    gemm(ws + O_Z512, 512, wbf + W_TPROJ, F(118), ws + O_HT, 256, NN, 512, 256, stream);
    ln(ws + O_HT, nullptr, F(117), F(116), ws + O_HT, NN, 256, 0, 0, stream);

    // ---- struct ----
    gemm(F(0), 128, wbf + W_INPROJ, F(48), ws + O_H, 256, NN, 128, 256, stream);
    fill(ws + O_CNT, 0.f, NN, stream);
    fill(ws + O_LAT, 0.f, (size_t)NN * 64, stream);
    k_edge_attr<<<EE * 64 / 256, 256, 0, stream>>>(I(1), I(7), F(43), ws + O_LAT, ws + O_CNT);
    k_loop_div<<<NN * 64 / 256, 256, 0, stream>>>(ws + O_LAT, ws + O_CNT);

    for (int l = 0; l < 3; l++) {
        int base = 50 + l * 20;
        fill(ws + O_HSUM, 0.f, (size_t)NN * 256, stream);
        for (int g = 0; g < 2; g++) {
            int gb = base + 2 + g * 7;
            size_t wb = W_LAYER + l * LAYER_SZ + g * GAT_SZ;
            gemm(ws + O_H, 256, wbf + wb + 64 * 256,             F(gb + 2), ws + O_XL, 256, NN, 256, 256, stream);
            gemm(ws + O_H, 256, wbf + wb + 64 * 256 + 256 * 256, F(gb + 3), ws + O_XR, 256, NN, 256, 256, stream);
            k_ee3<<<1, 256, 0, stream>>>(F(43), F(gb + 4), ws + O_EE3);
            gemm(ws + O_LAT, 64, wbf + wb, nullptr, ws + O_EEL, 256, NN, 64, 256, stream);
            fill(ws + O_NMAX, -1e30f, (size_t)NN * 4, stream);
            fill(ws + O_NSUM, 0.f, (size_t)NN * 4, stream);
            k_gat_score<<<ETOT / 8, 256, 0, stream>>>(I(1), I(7), ws + O_XL, ws + O_XR,
                                                      ws + O_EE3, ws + O_EEL, F(gb + 0),
                                                      ws + O_ESC, ws + O_NMAX);
            k_gat_expsum<<<ETOT * 4 / 256, 256, 0, stream>>>(I(1), ws + O_ESC,
                                                             ws + O_NMAX, ws + O_NSUM);
            k_gat_aggr<<<ETOT / 8, 256, 0, stream>>>(I(1), ws + O_XL, ws + O_ESC,
                                                     ws + O_NSUM, ws + O_HSUM);
        }
        k_comb_prep<<<NN * 320 / 256, 256, 0, stream>>>(ws + O_HSUM, F(base + 3), F(base + 10),
                                                        F(0), F(base + 19), F(base + 18),
                                                        ws + O_CAT);
        gemm(ws + O_CAT, 320, wbf + W_LAYER + l * LAYER_SZ + 2 * GAT_SZ, F(base + 0),
             ws + O_HNEW, 256, NN, 320, 256, stream);
        ln(ws + O_HNEW, ws + O_H, F(base + 17), F(base + 16), ws + O_H, NN, 256, 0,
           (l < 2) ? 1 : 0, stream);
    }

    // ---- fusion ----
    k_concat<<<NN * 768 / 256, 256, 0, stream>>>(ws + O_HS, ws + O_H, ws + O_HT, hc);
    gemm(hc, 768, wbf + W_F1, F(9), ws + O_Z512, 512, NN, 768, 512, stream);
    ln(ws + O_Z512, nullptr, F(16), F(15), ws + O_Z512, NN, 512, 1, 0, stream);
    gemm(ws + O_Z512, 512, wbf + W_F2, F(11), ws + O_Z256, 256, NN, 512, 256, stream);
    ln(ws + O_Z256, nullptr, F(18), F(17), ws + O_Z256, NN, 256, 1, 0, stream);
    k_scores<<<NN / 8, 256, 0, stream>>>(ws + O_Z256, F(14), F(13), out);
    k_value<<<BB, 256, 0, stream>>>(hc, F(122), F(121), F(124), F(123), vout);
#undef F
#undef I
}
